// GATModel_20117626814707
// MI455X (gfx1250) — matrix operand order repaired and checked by reading, not compiled, not run on hardware
//
#include <hip/hip_runtime.h>
#include <hip/hip_bf16.h>

// ---------------------------------------------------------------------------
// GAT (4 layers) + BN + MLP head for gfx1250 (MI455X).
// GEMMs: v_wmma_f32_16x16x32_bf16, 2x2 register-blocked (32x32 per wave).
// Edge softmax/scatter in f32 with atomics (bandwidth-bound anyway).
// ---------------------------------------------------------------------------

#define NNODES 50000
#define NPAD   50016           // NNODES rounded up to 32 for 2x2 WMMA blocking
#define NEDGES 800000
#define ETOT   (NEDGES + NNODES)
#define INCH   128
#define HIDC   64
#define CTMAX  256
#define HMAX   4
#define NEG_SLOPE 0.2f
#define BN_EPS 1e-5f

typedef __attribute__((ext_vector_type(16))) __bf16 v16bf;
typedef __attribute__((ext_vector_type(8)))  float  v8f;

__device__ __forceinline__ unsigned short f2bf(float f) {
    unsigned u = __float_as_uint(f);
    unsigned r = (u + 0x7FFFu + ((u >> 16) & 1u)) >> 16;   // RNE
    return (unsigned short)r;
}
__device__ __forceinline__ float bf2f(unsigned short us) {
    return __uint_as_float(((unsigned)us) << 16);
}
// order-preserving float<->uint for atomic max on signed floats
__device__ __forceinline__ unsigned fflip(float f) {
    unsigned u = __float_as_uint(f);
    return (u & 0x80000000u) ? ~u : (u | 0x80000000u);
}
__device__ __forceinline__ float funflip(unsigned u) {
    unsigned v = (u & 0x80000000u) ? (u ^ 0x80000000u) : ~u;
    return __uint_as_float(v);
}

// --- edge list with self loops, int32 ---------------------------------------
__global__ void k_prep_edges(const int* __restrict__ ei, int* __restrict__ src,
                             int* __restrict__ dst, int E, int et) {
    int i = blockIdx.x * blockDim.x + threadIdx.x;
    if (i >= et) return;
    if (i < E) { src[i] = ei[i]; dst[i] = ei[E + i]; }
    else       { src[i] = i - E; dst[i] = i - E; }
}

// --- f32 -> bf16 ------------------------------------------------------------
__global__ void k_f32_to_bf16(const float* __restrict__ in,
                              unsigned short* __restrict__ out, int n) {
    int i = blockIdx.x * blockDim.x + threadIdx.x;
    if (i < n) out[i] = f2bf(in[i]);
}

// --- W[K,Nt] f32 -> Wt[Nt,K] bf16 (column-major staging for WMMA B) --------
__global__ void k_wt(const float* __restrict__ W, unsigned short* __restrict__ wt,
                     int K, int Nt) {
    int t = blockIdx.x * blockDim.x + threadIdx.x;
    if (t >= K * Nt) return;
    int k = t / Nt, n = t % Nt;
    wt[(size_t)n * K + k] = f2bf(W[t]);
}

// --- bf16 WMMA GEMM: C[M,Nt] = A[M,K] * Bt[Nt,K]^T --------------------------
// 2x2 register-blocked: each wave produces a 32x32 output block (4 WMMA per
// 8 b128 loads per K-step). Control flow wave-uniform until the guarded
// epilogue stores, so EXEC is all-ones for every WMMA. A rows are padded to
// a multiple of 32 inside the workspace.
__global__ void k_gemm_wmma(const unsigned short* __restrict__ A,
                            const unsigned short* __restrict__ Bt,
                            float* __restrict__ C, int M, int K, int Nt) {
    int wave = (blockIdx.x * blockDim.x + threadIdx.x) >> 5;
    int lane = threadIdx.x & 31;
    int tilesN = Nt >> 5;
    int tilesM = (M + 31) >> 5;
    if (wave >= tilesM * tilesN) return;
    int tm = wave / tilesN, tn = wave % tilesN;
    int m16 = lane & 15, half = lane >> 4;

    const unsigned short* arow0 = A + (size_t)(tm * 32 + m16) * K;
    const unsigned short* arow1 = arow0 + (size_t)16 * K;
    const unsigned short* brow0 = Bt + (size_t)(tn * 32 + m16) * K;
    const unsigned short* brow1 = brow0 + (size_t)16 * K;

    v8f acc00 = {}, acc01 = {}, acc10 = {}, acc11 = {};
    for (int k0 = 0; k0 < K; k0 += 32) {
        __builtin_prefetch(arow0 + k0 + 64, 0, 1);   // global_prefetch_b8
        __builtin_prefetch(arow1 + k0 + 64, 0, 1);
        union { uint4 u[2]; v16bf v; } a0, a1, b0, b1;
        a0.u[0] = *(const uint4*)(arow0 + k0 +      half * 8);
        a0.u[1] = *(const uint4*)(arow0 + k0 + 16 + half * 8);
        a1.u[0] = *(const uint4*)(arow1 + k0 +      half * 8);
        a1.u[1] = *(const uint4*)(arow1 + k0 + 16 + half * 8);
        b0.u[0] = *(const uint4*)(brow0 + k0 +      half * 8);
        b0.u[1] = *(const uint4*)(brow0 + k0 + 16 + half * 8);
        b1.u[0] = *(const uint4*)(brow1 + k0 +      half * 8);
        b1.u[1] = *(const uint4*)(brow1 + k0 + 16 + half * 8);
        acc00 = __builtin_amdgcn_wmma_f32_16x16x32_bf16(false, a0.v, false, b0.v, (short)0, acc00, false, false);
        acc01 = __builtin_amdgcn_wmma_f32_16x16x32_bf16(false, a0.v, false, b1.v, (short)0, acc01, false, false);
        acc10 = __builtin_amdgcn_wmma_f32_16x16x32_bf16(false, a1.v, false, b0.v, (short)0, acc10, false, false);
        acc11 = __builtin_amdgcn_wmma_f32_16x16x32_bf16(false, a1.v, false, b1.v, (short)0, acc11, false, false);
    }
    int col = tn * 32 + m16;
#pragma unroll
    for (int r = 0; r < 8; ++r) {
        int row0 = tm * 32 + r + 8 * half;
        int row1 = row0 + 16;
        if (row0 < M) {
            C[(size_t)row0 * Nt + col]      = acc00[r];
            C[(size_t)row0 * Nt + col + 16] = acc01[r];
        }
        if (row1 < M) {
            C[(size_t)row1 * Nt + col]      = acc10[r];
            C[(size_t)row1 * Nt + col + 16] = acc11[r];
        }
    }
}

// --- per-node attention logits: asrc[n,h] = sum_c h[n,h,c]*a_src[h,c] -------
__global__ void k_attn(const float* __restrict__ hf, const float* __restrict__ As,
                       const float* __restrict__ Ad, float* __restrict__ asrc,
                       float* __restrict__ adst, int n, int H, int C) {
    int t = blockIdx.x * blockDim.x + threadIdx.x;
    if (t >= n * H) return;
    int node = t / H, h = t % H;
    const float* row = hf + (size_t)node * H * C + h * C;
    float s0 = 0.f, s1 = 0.f;
    for (int c = 0; c < C; ++c) { s0 += row[c] * As[h * C + c]; s1 += row[c] * Ad[h * C + c]; }
    asrc[t] = s0; adst[t] = s1;
}

// --- segment max over incoming edges (flipped-uint atomicMax) ---------------
__global__ void k_edge_max(const int* __restrict__ src, const int* __restrict__ dst,
                           const float* __restrict__ asrc, const float* __restrict__ adst,
                           unsigned* __restrict__ mflip, int et, int H) {
    int e = blockIdx.x * blockDim.x + threadIdx.x;
    if (e >= et) return;
    int s = src[e], d = dst[e];
    for (int h = 0; h < H; ++h) {
        float v = asrc[s * H + h] + adst[d * H + h];
        v = (v > 0.f) ? v : NEG_SLOPE * v;
        atomicMax(&mflip[d * H + h], fflip(v));
    }
}

// --- exp(e - max) + segment sum --------------------------------------------
__global__ void k_edge_expsum(const int* __restrict__ src, const int* __restrict__ dst,
                              const float* __restrict__ asrc, const float* __restrict__ adst,
                              const unsigned* __restrict__ mflip, float* __restrict__ exbuf,
                              float* __restrict__ esum, int et, int H) {
    int e = blockIdx.x * blockDim.x + threadIdx.x;
    if (e >= et) return;
    int s = src[e], d = dst[e];
    for (int h = 0; h < H; ++h) {
        float v = asrc[s * H + h] + adst[d * H + h];
        v = (v > 0.f) ? v : NEG_SLOPE * v;
        float m = funflip(mflip[d * H + h]);
        float ex = __expf(v - m);
        exbuf[(size_t)e * H + h] = ex;
        atomicAdd(&esum[d * H + h], ex);
    }
}

// --- weighted message scatter: acc[dst,h,:] += alpha * h[src,h,:] -----------
__global__ void k_edge_scatter(const int* __restrict__ src, const int* __restrict__ dst,
                               const float* __restrict__ hf, const float* __restrict__ exbuf,
                               const float* __restrict__ esum, float* __restrict__ acc,
                               int et, int H, int C) {
    int t = blockIdx.x * blockDim.x + threadIdx.x;
    if (t >= et * H) return;
    int e = t / H, h = t % H;
    int s = src[e], d = dst[e];
    float alpha = exbuf[(size_t)e * H + h] / (esum[d * H + h] + 1e-16f);
    const float* srow = hf  + (size_t)s * H * C + h * C;
    float*       drow = acc + (size_t)d * H * C + h * C;
    for (int c = 0; c < C; ++c) atomicAdd(&drow[c], srow[c] * alpha);
}

// --- BN reduce: per-channel sum/sumsq (blockDim == CT) ----------------------
__global__ void k_bn_reduce(const float* __restrict__ acc, float* __restrict__ s,
                            float* __restrict__ ss, int n, int CT, int chunk) {
    int c = threadIdx.x;
    int n0 = blockIdx.x * chunk;
    int n1 = min(n0 + chunk, n);
    float a = 0.f, b = 0.f;
    for (int i = n0; i < n1; ++i) {
        float v = acc[(size_t)i * CT + c];
        a += v; b += v * v;
    }
    atomicAdd(&s[c], a); atomicAdd(&ss[c], b);
}

// --- BN normalize + ReLU -> bf16 for next GEMM ------------------------------
__global__ void k_bn_norm(const float* __restrict__ acc, const float* __restrict__ s,
                          const float* __restrict__ ss, const float* __restrict__ g,
                          const float* __restrict__ be, unsigned short* __restrict__ xbf,
                          int n, int CT) {
    int t = blockIdx.x * blockDim.x + threadIdx.x;
    if (t >= n * CT) return;
    int c = t % CT;
    float mu  = s[c] / (float)n;
    float var = ss[c] / (float)n - mu * mu;
    float y = g[c] * (acc[t] - mu) * rsqrtf(var + BN_EPS) + be[c];
    xbf[t] = f2bf(fmaxf(y, 0.f));
}

// --- MLP head: 64 -> 32 (ReLU) -> 2 -----------------------------------------
__global__ void k_mlp(const unsigned short* __restrict__ xbf,
                      const float* __restrict__ hw1, const float* __restrict__ hb1,
                      const float* __restrict__ hw2, const float* __restrict__ hb2,
                      float* __restrict__ out, int n) {
    int i = blockIdx.x * blockDim.x + threadIdx.x;
    if (i >= n) return;
    float f[64];
#pragma unroll
    for (int c = 0; c < 64; ++c) f[c] = bf2f(xbf[(size_t)i * 64 + c]);
    float o0 = hb2[0], o1 = hb2[1];
    for (int j = 0; j < 32; ++j) {
        float hsum = hb1[j];
#pragma unroll
        for (int c = 0; c < 64; ++c) hsum += f[c] * hw1[c * 32 + j];
        hsum = fmaxf(hsum, 0.f);
        o0 += hsum * hw2[j * 2 + 0];
        o1 += hsum * hw2[j * 2 + 1];
    }
    out[(size_t)i * 2 + 0] = o0;
    out[(size_t)i * 2 + 1] = o1;
}

extern "C" void kernel_launch(void* const* d_in, const int* in_sizes, int n_in,
                              void* d_out, int out_size, void* d_ws, size_t ws_size,
                              hipStream_t stream) {
    (void)in_sizes; (void)n_in; (void)out_size; (void)ws_size;
    const float* x  = (const float*)d_in[0];
    const int*   ei = (const int*)d_in[1];
    const float* hw1 = (const float*)d_in[26];
    const float* hb1 = (const float*)d_in[27];
    const float* hw2 = (const float*)d_in[28];
    const float* hb2 = (const float*)d_in[29];
    float* out = (float*)d_out;

    // workspace layout
    char* ws = (char*)d_ws;
    size_t off = 0;
    auto nxt = [&](size_t bytes) { char* p = ws + off; off = (off + bytes + 255) & ~(size_t)255; return p; };
    int*            p_src = (int*)nxt((size_t)ETOT * 4);
    int*            p_dst = (int*)nxt((size_t)ETOT * 4);
    unsigned short* p_xbf = (unsigned short*)nxt((size_t)NPAD * CTMAX * 2);   // padded rows
    unsigned short* p_wt  = (unsigned short*)nxt((size_t)CTMAX * CTMAX * 2);
    float*          p_hf  = (float*)nxt((size_t)NNODES * CTMAX * 4);
    float*          p_as  = (float*)nxt((size_t)NNODES * HMAX * 4);
    float*          p_ad  = (float*)nxt((size_t)NNODES * HMAX * 4);
    unsigned*       p_mf  = (unsigned*)nxt((size_t)NNODES * HMAX * 4);
    float*          p_es  = (float*)nxt((size_t)NNODES * HMAX * 4);
    float*          p_ex  = (float*)nxt((size_t)ETOT * HMAX * 4);
    float*          p_acc = (float*)nxt((size_t)NNODES * CTMAX * 4);
    float*          p_s   = (float*)nxt((size_t)CTMAX * 4);
    float*          p_ss  = (float*)nxt((size_t)CTMAX * 4);

    auto cdiv = [](long long a, long long b) { return (int)((a + b - 1) / b); };

    // edges with self loops (int32)
    k_prep_edges<<<cdiv(ETOT, 256), 256, 0, stream>>>(ei, p_src, p_dst, NEDGES, ETOT);
    // layer-0 input to bf16
    k_f32_to_bf16<<<cdiv((long long)NNODES * INCH, 256), 256, 0, stream>>>(x, p_xbf, NNODES * INCH);

    const int Kdim[4] = {128, 256, 256, 256};
    const int Hn[4]   = {4, 4, 4, 1};
    const int CTn[4]  = {256, 256, 256, 64};
    const int widx[4] = {2, 8, 14, 20};

    for (int L = 0; L < 4; ++L) {
        const int K = Kdim[L], H = Hn[L], CT = CTn[L], C = HIDC;
        const float* W  = (const float*)d_in[widx[L] + 0];
        const float* As = (const float*)d_in[widx[L] + 1];
        const float* Ad = (const float*)d_in[widx[L] + 2];
        const float* g  = (const float*)d_in[widx[L] + 4];
        const float* be = (const float*)d_in[widx[L] + 5];
        // (bias b{L} dropped: constant channel shift cancels inside BN)

        // W -> Wt bf16 (column-major for WMMA B)
        k_wt<<<cdiv((long long)K * CT, 256), 256, 0, stream>>>(W, p_wt, K, CT);

        // hfeat = x @ W  (bf16 WMMA, f32 acc, 32x32 per wave)
        int waves = ((NNODES + 31) / 32) * (CT / 32);
        k_gemm_wmma<<<cdiv(waves, 8), 256, 0, stream>>>(p_xbf, p_wt, p_hf, NNODES, K, CT);

        // attention logits
        k_attn<<<cdiv((long long)NNODES * H, 256), 256, 0, stream>>>(p_hf, As, Ad, p_as, p_ad, NNODES, H, C);

        // zero accumulators
        hipMemsetAsync(p_mf, 0, (size_t)NNODES * H * 4, stream);
        hipMemsetAsync(p_es, 0, (size_t)NNODES * H * 4, stream);
        hipMemsetAsync(p_acc, 0, (size_t)NNODES * CT * 4, stream);
        hipMemsetAsync(p_s, 0, (size_t)CT * 4, stream);
        hipMemsetAsync(p_ss, 0, (size_t)CT * 4, stream);

        // segment softmax + scatter
        k_edge_max<<<cdiv(ETOT, 256), 256, 0, stream>>>(p_src, p_dst, p_as, p_ad, p_mf, ETOT, H);
        k_edge_expsum<<<cdiv(ETOT, 256), 256, 0, stream>>>(p_src, p_dst, p_as, p_ad, p_mf, p_ex, p_es, ETOT, H);
        k_edge_scatter<<<cdiv((long long)ETOT * H, 256), 256, 0, stream>>>(p_src, p_dst, p_hf, p_ex, p_es, p_acc, ETOT, H, C);

        // BN + ReLU -> bf16 for next layer
        k_bn_reduce<<<cdiv(NNODES, 256), CT, 0, stream>>>(p_acc, p_s, p_ss, NNODES, CT, 256);
        k_bn_norm<<<cdiv((long long)NNODES * CT, 256), 256, 0, stream>>>(p_acc, p_s, p_ss, g, be, p_xbf, NNODES, CT);
    }

    // MLP head: [N,64] -> [N,2]
    k_mlp<<<cdiv(NNODES, 256), 256, 0, stream>>>(p_xbf, hw1, hb1, hw2, hb2, out, NNODES);
}
